// transformer_59081570124985
// MI455X (gfx1250) — matrix operand order repaired and checked by reading, not compiled, not run on hardware
//
#include <hip/hip_runtime.h>

// Shapes (compile-time, from the reference)
#define BQ 2
#define NQ 2048
#define CQ 512
#define HQ 8
#define DQ 64
#define FQ (3 * HQ * DQ)   // 1536
#define HD (HQ * DQ)       // 512

typedef __attribute__((ext_vector_type(16))) __bf16 v16bf;
typedef __attribute__((ext_vector_type(8)))  __bf16 v8bf;
typedef __attribute__((ext_vector_type(8)))  float  v8f;

typedef unsigned int u32x4 __attribute__((ext_vector_type(4)));
typedef int          i32x4 __attribute__((ext_vector_type(4)));
typedef int          i32x8 __attribute__((ext_vector_type(8)));

static __device__ __forceinline__ v8f wmma_bf16(v16bf a, v16bf b, v8f c) {
  // D = A(16x32 bf16) * B(32x16 bf16) + C(16x16 f32)
  return __builtin_amdgcn_wmma_f32_16x16x32_bf16(false, a, false, b, (short)0, c,
                                                 false, false);
}

// LDS byte offset of a __shared__ object (generic -> AS(3) cast folds to the
// static LDS allocation offset).
static __device__ __forceinline__ unsigned lds_offset_of(const void* p) {
  return (unsigned)(uintptr_t)(__attribute__((address_space(3))) const void*)p;
}

// ---- Tensor Data Mover: DMA a (tile_h x tile_w) f32 tile (row stride
// `stride` elements) from global memory into LDS.  D# layout per CDNA5 ISA
// CDNA5 ISA (group0: count/lds_addr/global_addr/type, group1:
// data_size/dims/tile/strides).  2D tensor -> groups 2/3 zero.
static __device__ __forceinline__ void tdm_load_f32_tile(unsigned lds_off,
                                                         const float* gptr,
                                                         int tile_w, int tile_h,
                                                         int stride) {
  unsigned long long ga = (unsigned long long)(uintptr_t)gptr;
  unsigned ga_lo = (unsigned)__builtin_amdgcn_readfirstlane((int)(ga & 0xffffffffu));
  unsigned ga_hi = (unsigned)__builtin_amdgcn_readfirstlane((int)((ga >> 32) & 0x01ffffffu));
  unsigned ldsa  = (unsigned)__builtin_amdgcn_readfirstlane((int)lds_off);

  u32x4 g0;
  g0[0] = 1u;                         // count=1, user flags 0
  g0[1] = ldsa;                       // lds_addr (bytes)
  g0[2] = ga_lo;                      // global_addr[31:0]
  g0[3] = ga_hi | (2u << 30);         // global_addr[56:32] | type=2 (image)

  i32x8 g1;
  g1[0] = (2 << 16);                  // workgroup_mask=0, data_size=2 (4B)
  g1[1] = (NQ & 0xffff) << 16;        // tensor_dim0[15:0] (=2048)
  g1[2] = ((NQ >> 16) & 0xffff) | ((NQ & 0xffff) << 16);   // dim0 hi | dim1 lo
  g1[3] = ((NQ >> 16) & 0xffff) | ((tile_w & 0xffff) << 16); // dim1 hi | tile_dim0
  g1[4] = (tile_h & 0xffff);          // tile_dim1 | tile_dim2=0
  g1[5] = stride;                     // tensor_dim0_stride[31:0]
  g1[6] = 0;                          // dim0_stride hi | dim1_stride lo
  g1[7] = 0;

  i32x4 z4 = {0, 0, 0, 0};
#if defined(__clang_major__) && (__clang_major__ >= 23)
  i32x8 z8 = {0, 0, 0, 0, 0, 0, 0, 0};
  __builtin_amdgcn_tensor_load_to_lds(g0, g1, z4, z4, z8, 0);
#else
  __builtin_amdgcn_tensor_load_to_lds(g0, g1, z4, z4, 0);
#endif
}

// ---- WMMA operand loaders (per CDNA5 ISA VGPR layout tables, wave32) ----
// A 16x32 (16-bit): lanes 0-15 row=lane, K elems {0..7,16..23}; lanes 16-31
// same row, K elems {8..15,24..31}.
static __device__ __forceinline__ v16bf load_a_f32(const float* src, int ld,
                                                   int row0, int k0, int lane) {
  const float* p = src + (size_t)(row0 + (lane & 15)) * ld + k0 + ((lane & 16) ? 8 : 0);
  v16bf a;
#pragma unroll
  for (int e = 0; e < 8; ++e) { a[e] = (__bf16)p[e]; a[8 + e] = (__bf16)p[16 + e]; }
  return a;
}

static __device__ __forceinline__ v16bf load_a_bf16(const __bf16* src, int ld,
                                                    int row0, int k0, int lane) {
  const __bf16* p = src + (size_t)(row0 + (lane & 15)) * ld + k0 + ((lane & 16) ? 8 : 0);
  v8bf lo = *(const v8bf*)p;          // 16B aligned
  v8bf hi = *(const v8bf*)(p + 16);
  v16bf a;
#pragma unroll
  for (int e = 0; e < 8; ++e) { a[e] = lo[e]; a[8 + e] = hi[e]; }
  return a;
}

// B 32x16 (16-bit): lane holds column n = lane&15; lanes 0-15 hold K {0..7,16..23},
// lanes 16-31 hold K {8..15,24..31}.  Transpose-on-load: element(k,n) =
// src[(n0+n)*ld + k0+k].
static __device__ __forceinline__ v16bf load_bT_f32(const float* src, int ld,
                                                    int n0, int k0, int lane) {
  const float* p = src + (size_t)(n0 + (lane & 15)) * ld + k0 + ((lane & 16) ? 8 : 0);
  v16bf b;
#pragma unroll
  for (int e = 0; e < 16; ++e) b[e] = (__bf16)p[e + (e & 8)];
  return b;
}

static __device__ __forceinline__ v16bf load_bT_bf16(const __bf16* src, int ld,
                                                     int n0, int k0, int lane) {
  const __bf16* p = src + (size_t)(n0 + (lane & 15)) * ld + k0 + ((lane & 16) ? 8 : 0);
  v8bf lo = *(const v8bf*)p;          // 16B aligned
  v8bf hi = *(const v8bf*)(p + 16);
  v16bf b;
#pragma unroll
  for (int e = 0; e < 8; ++e) { b[e] = lo[e]; b[8 + e] = hi[e]; }
  return b;
}

// K-major B: element(k,n) = src[(k0+k)*ld + n0+n]   (used for attn * V)
static __device__ __forceinline__ v16bf load_bK_bf16(const __bf16* src, int ld,
                                                     int k0, int n0, int lane) {
  int n = n0 + (lane & 15);
  int kb = k0 + ((lane & 16) ? 8 : 0);
  v16bf b;
#pragma unroll
  for (int e = 0; e < 16; ++e) b[e] = src[(size_t)(kb + e + (e & 8)) * ld + n];
  return b;
}

// ======================= Kernel 1: QKV projection =========================
// proj[b,n,f] = sum_c x[b,n,c] * Wqkv[f,c];  f = h*192 + d*3 + sel
// q is pre-scaled by D^-0.5 (legal fold: the Wpre mix is linear).
__global__ __launch_bounds__(256) void qkv_wmma(const float* __restrict__ x,
                                                const float* __restrict__ Wqkv,
                                                __bf16* __restrict__ qs,
                                                __bf16* __restrict__ ks,
                                                __bf16* __restrict__ vs) {
  int lane = threadIdx.x & 31;
  int wave = threadIdx.x >> 5;
  int tile = blockIdx.x * 8 + wave;
  const int tcols = FQ / 16;                 // 96
  int row0 = (tile / tcols) * 16;
  int col0 = (tile % tcols) * 16;

  v8f acc = {0, 0, 0, 0, 0, 0, 0, 0};
#pragma unroll 4
  for (int k0 = 0; k0 < CQ; k0 += 32) {
    v16bf a = load_a_f32(x, CQ, row0, k0, lane);
    v16bf b = load_bT_f32(Wqkv, CQ, col0, k0, lane);   // Wqkv[f,c] -> B[k=c][n=f]
    acc = wmma_bf16(a, b, acc);
  }

  int f = col0 + (lane & 15);
  int h = f / 192, rem = f % 192, d = rem / 3, sel = rem % 3;
  __bf16* dst = (sel == 0) ? qs : (sel == 1) ? ks : vs;
  float scale = (sel == 0) ? 0.125f : 1.0f;  // 64^-0.5
  int rbase = (lane & 16) ? 8 : 0;
#pragma unroll
  for (int r = 0; r < 8; ++r) {
    int row = row0 + rbase + r;
    int b_ = row >> 11, n = row & (NQ - 1);
    dst[(((size_t)b_ * HQ + h) * NQ + n) * DQ + d] = (__bf16)(acc[r] * scale);
  }
}

// ================== Kernel 2: fused talking-heads attention ===============
// One workgroup = 8 wave32 = 1 wave per head, per (b, 16-row query tile).
// Two sweeps over the 2048 keys (j-tiles of 32): sweep 1 accumulates softmax
// stats of the PRE-mixed logits; sweep 2 normalizes, POST-mixes in LDS, and
// accumulates attn''*V with WMMA.  pos_bias tiles are DMA'd into LDS by the
// Tensor Data Mover, overlapped with the Q*K^T WMMAs.
__global__ __launch_bounds__(256) void attn_wmma(const __bf16* __restrict__ qs,
                                                 const __bf16* __restrict__ ks,
                                                 const __bf16* __restrict__ vs,
                                                 const float* __restrict__ pos_bias,
                                                 const unsigned char* __restrict__ mask,
                                                 const float* __restrict__ Wpre,
                                                 const float* __restrict__ bpre,
                                                 const float* __restrict__ Wpost,
                                                 const float* __restrict__ bpost,
                                                 __bf16* __restrict__ oh) {
  __shared__ float Pl[HQ][16][32];   // raw / post-mixed tiles
  __shared__ float Ql[HQ][16][32];   // normalized attention tiles
  __shared__ float PB[HQ][16][32];   // TDM-staged pos_bias tiles (per head)

  int lane = threadIdx.x & 31;
  int w = threadIdx.x >> 5;                       // head
  int b = blockIdx.x >> 7;                        // 128 row-tiles per batch
  int i0 = (blockIdx.x & 127) * 16;

  const __bf16* qh = qs + (size_t)(b * HQ + w) * NQ * DQ;
  const __bf16* kh = ks + (size_t)(b * HQ + w) * NQ * DQ;
  const __bf16* vh = vs + (size_t)(b * HQ + w) * NQ * DQ;
  const float* pb = pos_bias + (size_t)w * NQ * NQ;
  const unsigned char* mk = mask + (size_t)b * NQ * NQ;

  v16bf qa0 = load_a_bf16(qh, DQ, i0, 0, lane);
  v16bf qa1 = load_a_bf16(qh, DQ, i0, 32, lane);

  float wpre[HQ], wpost[HQ];
#pragma unroll
  for (int h = 0; h < HQ; ++h) { wpre[h] = Wpre[w * HQ + h]; wpost[h] = Wpost[w * HQ + h]; }
  float bpre_w = bpre[w], bpost_w = bpost[w];

  int rr = (lane & 16) ? 8 : 0;
  int cl = lane & 15;
  unsigned pb_lds = lds_offset_of(&PB[w][0][0]);
  const v8f z = {0, 0, 0, 0, 0, 0, 0, 0};

  float m[8], l[8];
#pragma unroll
  for (int r = 0; r < 8; ++r) { m[r] = -3.0e38f; l[r] = 0.0f; }

  // ---------------- sweep 1: softmax statistics ----------------
  for (int j0 = 0; j0 < NQ; j0 += 32) {
    // async: DMA this head's 16x32 pos_bias tile into LDS
    tdm_load_f32_tile(pb_lds, pb + (size_t)i0 * NQ + j0, 32, 16, NQ);
    // prefetch next j-tile of K (one 128B row per lane covers rows j0+32..63)
    if (j0 + 32 < NQ) __builtin_prefetch(kh + (size_t)(j0 + 32 + lane) * DQ, 0, 0);

    v8f c0 = wmma_bf16(qa0, load_bT_bf16(kh, DQ, j0, 0, lane), z);
    c0 = wmma_bf16(qa1, load_bT_bf16(kh, DQ, j0, 32, lane), c0);
    v8f c1 = wmma_bf16(qa0, load_bT_bf16(kh, DQ, j0 + 16, 0, lane), z);
    c1 = wmma_bf16(qa1, load_bT_bf16(kh, DQ, j0 + 16, 32, lane), c1);
#pragma unroll
    for (int r = 0; r < 8; ++r) { Pl[w][rr + r][cl] = c0[r]; Pl[w][rr + r][16 + cl] = c1[r]; }
    __builtin_amdgcn_s_wait_tensorcnt(0);
    __syncthreads();
#pragma unroll
    for (int r = 0; r < 8; ++r) {
      int row = rr + r, i = i0 + row;
#pragma unroll
      for (int half = 0; half < 2; ++half) {
        int col = half * 16 + cl, j = j0 + col;
        float s = bpre_w;
#pragma unroll
        for (int h = 0; h < HQ; ++h) s += wpre[h] * Pl[h][row][col];
        s += PB[w][row][col];
        if (mk[(size_t)i * NQ + j]) s = -3.0e38f;
        float mn = fmaxf(m[r], s);
        l[r] = l[r] * __expf(m[r] - mn) + __expf(s - mn);
        m[r] = mn;
      }
    }
    __syncthreads();
  }
  // combine partial (m,l) across the 16 lanes holding each row
#pragma unroll
  for (int off = 1; off < 16; off <<= 1) {
#pragma unroll
    for (int r = 0; r < 8; ++r) {
      float m2 = __shfl_xor(m[r], off, 32);
      float l2 = __shfl_xor(l[r], off, 32);
      float mn = fmaxf(m[r], m2);
      l[r] = l[r] * __expf(m[r] - mn) + l2 * __expf(m2 - mn);
      m[r] = mn;
    }
  }
  float linv[8];
#pragma unroll
  for (int r = 0; r < 8; ++r) linv[r] = 1.0f / l[r];

  // ---------------- sweep 2: normalize, post-mix, attn * V -------------
  v8f acc[4];
#pragma unroll
  for (int t = 0; t < 4; ++t) acc[t] = z;

  for (int j0 = 0; j0 < NQ; j0 += 32) {
    tdm_load_f32_tile(pb_lds, pb + (size_t)i0 * NQ + j0, 32, 16, NQ);
    if (j0 + 32 < NQ) {
      __builtin_prefetch(kh + (size_t)(j0 + 32 + lane) * DQ, 0, 0);
      __builtin_prefetch(vh + (size_t)(j0 + 32 + lane) * DQ, 0, 0);
    }

    v8f c0 = wmma_bf16(qa0, load_bT_bf16(kh, DQ, j0, 0, lane), z);
    c0 = wmma_bf16(qa1, load_bT_bf16(kh, DQ, j0, 32, lane), c0);
    v8f c1 = wmma_bf16(qa0, load_bT_bf16(kh, DQ, j0 + 16, 0, lane), z);
    c1 = wmma_bf16(qa1, load_bT_bf16(kh, DQ, j0 + 16, 32, lane), c1);
#pragma unroll
    for (int r = 0; r < 8; ++r) { Pl[w][rr + r][cl] = c0[r]; Pl[w][rr + r][16 + cl] = c1[r]; }
    __builtin_amdgcn_s_wait_tensorcnt(0);
    __syncthreads();
    // pre-mix + biases + mask, normalize -> Ql
#pragma unroll
    for (int r = 0; r < 8; ++r) {
      int row = rr + r, i = i0 + row;
#pragma unroll
      for (int half = 0; half < 2; ++half) {
        int col = half * 16 + cl, j = j0 + col;
        float s = bpre_w;
#pragma unroll
        for (int h = 0; h < HQ; ++h) s += wpre[h] * Pl[h][row][col];
        s += PB[w][row][col];
        if (mk[(size_t)i * NQ + j]) s = -3.0e38f;
        Ql[w][row][col] = __expf(s - m[r]) * linv[r];
      }
    }
    __syncthreads();
    // post talking-heads mix -> back into own Pl[w]
#pragma unroll
    for (int r = 0; r < 8; ++r) {
      int row = rr + r;
#pragma unroll
      for (int half = 0; half < 2; ++half) {
        int col = half * 16 + cl;
        float a = bpost_w;
#pragma unroll
        for (int h = 0; h < HQ; ++h) a += wpost[h] * Ql[h][row][col];
        Pl[w][row][col] = a;
      }
    }
    // same-wave LDS write->read (DS ops are in-order within a wave; the
    // compiler inserts s_wait_dscnt before the dependent loads)
    v16bf aA;
    {
      int arow = lane & 15;
      int kb = (lane & 16) ? 8 : 0;
#pragma unroll
      for (int e = 0; e < 8; ++e) {
        aA[e] = (__bf16)Pl[w][arow][kb + e];
        aA[8 + e] = (__bf16)Pl[w][arow][kb + 16 + e];
      }
    }
#pragma unroll
    for (int t = 0; t < 4; ++t)
      acc[t] = wmma_bf16(aA, load_bK_bf16(vh, DQ, j0, t * 16, lane), acc[t]);
  }

  // store per-head output [B,H,N,D] bf16
#pragma unroll
  for (int t = 0; t < 4; ++t)
#pragma unroll
    for (int r = 0; r < 8; ++r) {
      int i = i0 + rr + r, d = t * 16 + cl;
      oh[((size_t)(b * HQ + w) * NQ + i) * DQ + d] = (__bf16)acc[t][r];
    }
}

// ==================== Kernel 3: output projection =========================
// out[b,n,c] = sum_f oh[b, f/64, n, f%64] * Wout[c,f]
__global__ __launch_bounds__(256) void out_wmma(const __bf16* __restrict__ oh,
                                                const float* __restrict__ Wout,
                                                float* __restrict__ out) {
  int lane = threadIdx.x & 31;
  int wave = threadIdx.x >> 5;
  int tile = blockIdx.x * 8 + wave;
  const int tcols = CQ / 16;                 // 32
  int row0 = (tile / tcols) * 16;
  int col0 = (tile % tcols) * 16;

  v8f acc = {0, 0, 0, 0, 0, 0, 0, 0};
  int grow = row0 + (lane & 15);
  int b_ = grow >> 11, n = grow & (NQ - 1);
#pragma unroll 4
  for (int k0 = 0; k0 < HD; k0 += 32) {
    v16bf a;
    int kb = k0 + ((lane & 16) ? 8 : 0);
#pragma unroll
    for (int e = 0; e < 8; ++e) {
      int f0 = kb + e, f1 = kb + 16 + e;
      a[e]     = oh[(((size_t)b_ * HQ + (f0 >> 6)) * NQ + n) * DQ + (f0 & 63)];
      a[8 + e] = oh[(((size_t)b_ * HQ + (f1 >> 6)) * NQ + n) * DQ + (f1 & 63)];
    }
    v16bf bt = load_bT_f32(Wout, HD, col0, k0, lane);  // Wout[c,f] -> B[k=f][n=c]
    acc = wmma_bf16(a, bt, acc);
  }

  int c = col0 + (lane & 15);
  int rbase = (lane & 16) ? 8 : 0;
#pragma unroll
  for (int r = 0; r < 8; ++r)
    out[(size_t)(row0 + rbase + r) * CQ + c] = acc[r];
}

// ============================ launcher ====================================
extern "C" void kernel_launch(void* const* d_in, const int* in_sizes, int n_in,
                              void* d_out, int out_size, void* d_ws, size_t ws_size,
                              hipStream_t stream) {
  const float* x            = (const float*)d_in[0];
  const float* pos_bias     = (const float*)d_in[1];
  const unsigned char* mask = (const unsigned char*)d_in[2];
  const float* Wqkv         = (const float*)d_in[3];
  const float* Wout         = (const float*)d_in[4];
  const float* Wpre         = (const float*)d_in[5];
  const float* bpre         = (const float*)d_in[6];
  const float* Wpost        = (const float*)d_in[7];
  const float* bpost        = (const float*)d_in[8];
  float* out = (float*)d_out;

  const size_t nper = (size_t)BQ * HQ * NQ * DQ;       // 2,097,152 elems
  char* ws = (char*)d_ws;
  __bf16* qs = (__bf16*)(ws + 0 * nper * 2);
  __bf16* ks = (__bf16*)(ws + 1 * nper * 2);
  __bf16* vs = (__bf16*)(ws + 2 * nper * 2);
  __bf16* oh = (__bf16*)(ws + 3 * nper * 2);           // 16 MB total

  // QKV GEMM: (B*N/16)*(F/16) = 256*96 = 24576 wave-tiles, 8 waves/block
  qkv_wmma<<<(BQ * NQ / 16) * (FQ / 16) / 8, 256, 0, stream>>>(x, Wqkv, qs, ks, vs);
  // Fused talking-heads attention: one block per (b, 16-row query tile)
  attn_wmma<<<BQ * (NQ / 16), 256, 0, stream>>>(qs, ks, vs, pos_bias, mask,
                                                Wpre, bpre, Wpost, bpost, oh);
  // Output GEMM: (B*N/16)*(C/16) = 256*32 = 8192 wave-tiles, 8 waves/block
  out_wmma<<<(BQ * NQ / 16) * (CQ / 16) / 8, 256, 0, stream>>>(oh, Wout, out);

  (void)in_sizes; (void)n_in; (void)out_size; (void)ws_size;
}
